// MambaModel_74208444940359
// MI455X (gfx1250) — matrix operand order repaired and checked by reading, not compiled, not run on hardware
//
#include <hip/hip_runtime.h>

// ---------------------------------------------------------------------------
// MI455X (gfx1250) Mamba-style model forward, bf16 WMMA everywhere.
//  * all GEMMs: out[m,n] = sum_k A[m,k]*W[n,k], A/W pre-converted to bf16
//  * weights+embed converted f32->bf16 ONCE per launch (weight scratch lives
//    in d_out, which the head GEMM fully rewrites at the end)
//  * GEMM stages tiles with GLOBAL_LOAD_ASYNC_TO_LDS_B128 (ASYNCcnt) when the
//    builtin exists, double-buffered; plain copies otherwise
// ---------------------------------------------------------------------------

typedef __attribute__((ext_vector_type(16))) __bf16 v16bf;
typedef __attribute__((ext_vector_type(8)))  float  v8f;
typedef int v4i __attribute__((vector_size(16)));   // matches builtin param type

#define AS1 __attribute__((address_space(1)))
#define AS3 __attribute__((address_space(3)))

#if defined(__HIP_DEVICE_COMPILE__) && __has_builtin(__builtin_amdgcn_global_load_async_to_lds_b128)
#define USE_ASYNC 1
#else
#define USE_ASYNC 0
#endif

#if USE_ASYNC && __has_builtin(__builtin_amdgcn_s_wait_asynccnt)
#define WAIT_ASYNC() __builtin_amdgcn_s_wait_asynccnt(0)
#elif USE_ASYNC
#define WAIT_ASYNC() asm volatile("s_wait_asynccnt 0x0" ::: "memory")
#else
#define WAIT_ASYNC() do {} while (0)
#endif

struct Frag32B { uint4 lo, hi; };

static __device__ __forceinline__ unsigned short f2bf(float f) {
  __bf16 h = (__bf16)f;                      // native v_cvt
  return __builtin_bit_cast(unsigned short, h);
}
static __device__ __forceinline__ unsigned pack2(float a, float b) {
  return (unsigned)f2bf(a) | ((unsigned)f2bf(b) << 16);
}

// 16B global(bf16) -> LDS copy, async when available
static __device__ __forceinline__ void cp16_g2l(unsigned* l, const unsigned short* g) {
#if USE_ASYNC
  __builtin_amdgcn_global_load_async_to_lds_b128(
      (AS1 v4i*)(uintptr_t)g, (AS3 v4i*)(unsigned)(uintptr_t)l, 0, 0);
#else
  *(uint4*)l = *(const uint4*)g;
#endif
}

#define GEMM_THREADS 128
#define LDSW 20                 // uints per 32-bf16 row (16 data + 4 pad = 80B)
#define TILEU (64 * LDSW)       // uints per 64x32 tile

// OMODE: 0 = f32 out, 1 = bf16 out, 2 = both.  ACT: 0 none, 1 silu, 2 softplus
template<int OMODE, int ACT, bool HAS_BIAS, bool HAS_ADD>
__global__ __launch_bounds__(GEMM_THREADS)
void gemm_bf16(const unsigned short* __restrict__ Abf,  // [M,K] bf16
               const unsigned short* __restrict__ Wbf,  // [N,K] bf16
               const float* __restrict__ bias,
               const float* __restrict__ addsrc,
               float* __restrict__ outf,
               unsigned short* __restrict__ outb,
               int M, int N, int K)
{
  __shared__ unsigned lds[2 * 2 * TILEU];    // [buf][A|W][64*LDSW]  (20 KB)

  const int tid  = threadIdx.x;
  const int lane = tid & 31;
  const int wv   = tid >> 5;
  const int wm   = (wv >> 1) * 32;
  const int wn   = (wv & 1)  * 32;
  const int m0   = blockIdx.y * 64;
  const int n0   = blockIdx.x * 64;
  const int lr   = lane & 15;
  const int half = lane >> 4;

  // stage 64x32 bf16 tiles of A and W into LDS buffer `buf`
  auto stage = [&](int kb, int buf) {
    unsigned* dA = &lds[buf * 2 * TILEU];
    unsigned* dW = dA + TILEU;
#pragma unroll
    for (int i = 0; i < 2; ++i) {
      int q   = tid + i * GEMM_THREADS;  // 0..255: (row, 16B-chunk)
      int row = q >> 2;
      int c   = q & 3;
      {
        int gm = m0 + row; if (gm >= M) gm = M - 1;      // clamp: no OOB fault
        cp16_g2l(dA + row * LDSW + c * 4, Abf + (size_t)gm * K + kb + c * 8);
      }
      {
        int gn = n0 + row; if (gn >= N) gn = N - 1;
        cp16_g2l(dW + row * LDSW + c * 4, Wbf + (size_t)gn * K + kb + c * 8);
      }
    }
  };

  v8f acc[2][2] = {};
  const int nk = K >> 5;

  stage(0, 0);
  for (int kt = 0; kt < nk; ++kt) {
    const int buf = kt & 1;
    WAIT_ASYNC();
    __syncthreads();
    if (kt + 1 < nk) stage((kt + 1) << 5, buf ^ 1);

    const unsigned* lA = &lds[buf * 2 * TILEU];
    const unsigned* lW = lA + TILEU;

    // A 16x32 bf16 fragment: lane-half h holds K in {8h..8h+7} U {16+8h..+7}
    v16bf af[2], wf[2];
#pragma unroll
    for (int i = 0; i < 2; ++i) {
      const unsigned* p = &lA[(wm + i * 16 + lr) * LDSW + half * 4];
      Frag32B t; t.lo = *(const uint4*)p; t.hi = *(const uint4*)(p + 8);
      af[i] = __builtin_bit_cast(v16bf, t);
    }
    // B 32x16 bf16 fragment: lane-half h holds K in {8h..8h+7} U {16+8h..+7}
#pragma unroll
    for (int j = 0; j < 2; ++j) {
      const unsigned* p = &lW[(wn + j * 16 + lr) * LDSW + half * 4];
      Frag32B t; t.lo = *(const uint4*)p; t.hi = *(const uint4*)(p + 8);
      wf[j] = __builtin_bit_cast(v16bf, t);
    }
#pragma unroll
    for (int i = 0; i < 2; ++i)
#pragma unroll
      for (int j = 0; j < 2; ++j)
        acc[i][j] = __builtin_amdgcn_wmma_f32_16x16x32_bf16(
            false, af[i], false, wf[j], (short)0, acc[i][j], false, false);
    __syncthreads();
  }

  // epilogue: D element (m = half*8 + r, n = lr) per fragment
#pragma unroll
  for (int j = 0; j < 2; ++j) {
    int n = n0 + wn + j * 16 + lr;
    if (n >= N) continue;
    float bv = HAS_BIAS ? bias[n] : 0.f;
#pragma unroll
    for (int i = 0; i < 2; ++i) {
#pragma unroll
      for (int r = 0; r < 8; ++r) {
        int m = m0 + wm + i * 16 + half * 8 + r;
        if (m >= M) continue;
        float v = acc[i][j][r] + bv;
        if (ACT == 1) v = v / (1.f + __expf(-v));               // SiLU
        if (ACT == 2) v = (v > 20.f) ? v : log1pf(__expf(v));   // softplus
        size_t o = (size_t)m * N + n;
        if (HAS_ADD) v += addsrc[o];
        if (OMODE != 1) outf[o] = v;
        if (OMODE != 0) outb[o] = f2bf(v);
      }
    }
  }
}

// ---------------------------------------------------------------------------

__global__ void cvt_f32_bf16(const float* __restrict__ in,
                             unsigned short* __restrict__ out, size_t n)
{
  size_t i = ((size_t)blockIdx.x * blockDim.x + threadIdx.x) * 4;
  if (i + 3 < n) {
    float4 v = *(const float4*)(in + i);
    *(uint2*)(out + i) = make_uint2(pack2(v.x, v.y), pack2(v.z, v.w));
  } else {
    for (; i < n; ++i) out[i] = f2bf(in[i]);
  }
}

__global__ void embed_gather_bf(const int* __restrict__ tok,
                                const unsigned short* __restrict__ emb,
                                unsigned short* __restrict__ x, int rows, int d)
{
  int idx = blockIdx.x * blockDim.x + threadIdx.x;
  if (idx >= rows * d) return;
  int r = idx / d, e = idx - r * d;
  x[idx] = emb[(size_t)tok[r] * d + e];
}

// depthwise conv1d (k=3, pad=1 along L) + bias + SiLU -> bf16
__global__ void conv_silu(const float* __restrict__ hin, const float* __restrict__ cw,
                          const float* __restrict__ cb, unsigned short* __restrict__ hout,
                          int L, int d, int total)
{
  int idx = blockIdx.x * blockDim.x + threadIdx.x;
  if (idx >= total) return;
  int e = idx % d;
  int l = (idx / d) % L;
  float w0 = cw[e * 3 + 0], w1 = cw[e * 3 + 1], w2 = cw[e * 3 + 2];
  float xm = (l > 0)     ? hin[idx - d] : 0.f;
  float xc = hin[idx];
  float xp = (l < L - 1) ? hin[idx + d] : 0.f;
  float v = xm * w0 + xc * w1 + xp * w2 + cb[e];
  hout[idx] = f2bf(v / (1.f + __expf(-v)));
}

// SSM scan: c *= exp(-delta*A); acc += Bx*c  (sequential in L) -> bf16 hs
__global__ void ssm_scan(const float* __restrict__ delta, const float* __restrict__ At,
                         const float* __restrict__ Bx, unsigned short* __restrict__ hs,
                         int B, int L, int dh)
{
  int tid = blockIdx.x * blockDim.x + threadIdx.x;
  if (tid >= B * dh) return;
  int b = tid / dh, ch = tid - b * dh;
  size_t base = (size_t)b * L * dh + ch;
  float c = 1.f, acc = 0.f;
  for (int l = 0; l < L; ++l) {
    size_t o = base + (size_t)l * dh;
    c *= __expf(-delta[o] * At[o]);
    acc += Bx[o] * c;
    hs[o] = f2bf(acc);
  }
}

__global__ void mul_ew_bf(const float* __restrict__ a, const float* __restrict__ b,
                          unsigned short* __restrict__ o, int n)
{
  int i = blockIdx.x * blockDim.x + threadIdx.x;
  if (i < n) o[i] = f2bf(a[i] * b[i]);
}

__global__ __launch_bounds__(256)
void layernorm_bf(const float* __restrict__ x, const float* __restrict__ g,
                  const float* __restrict__ b, unsigned short* __restrict__ y, int d)
{
  __shared__ float sm[256];
  const int row = blockIdx.x;
  const float* xr = x + (size_t)row * d;
  float s = 0.f;
  for (int i = threadIdx.x; i < d; i += 256) s += xr[i];
  sm[threadIdx.x] = s; __syncthreads();
  for (int st = 128; st > 0; st >>= 1) {
    if (threadIdx.x < st) sm[threadIdx.x] += sm[threadIdx.x + st];
    __syncthreads();
  }
  const float mu = sm[0] / (float)d;
  __syncthreads();
  float v = 0.f;
  for (int i = threadIdx.x; i < d; i += 256) { float t = xr[i] - mu; v += t * t; }
  sm[threadIdx.x] = v; __syncthreads();
  for (int st = 128; st > 0; st >>= 1) {
    if (threadIdx.x < st) sm[threadIdx.x] += sm[threadIdx.x + st];
    __syncthreads();
  }
  const float inv = rsqrtf(sm[0] / (float)d + 1e-5f);
  for (int i = threadIdx.x; i < d; i += 256)
    y[(size_t)row * d + i] = f2bf((xr[i] - mu) * inv * g[i] + b[i]);
}

// ---------------------------------------------------------------------------

extern "C" void kernel_launch(void* const* d_in, const int* in_sizes, int n_in,
                              void* d_out, int out_size, void* d_ws, size_t ws_size,
                              hipStream_t stream)
{
  (void)in_sizes; (void)n_in; (void)out_size; (void)ws_size;

  const int*   tokens = (const int*)  d_in[0];
  const float* embedw = (const float*)d_in[1];
  const float* lin1_w = (const float*)d_in[2];
  const float* conv_w = (const float*)d_in[3];
  const float* conv_b = (const float*)d_in[4];
  const float* A_w    = (const float*)d_in[5];
  const float* A_b    = (const float*)d_in[6];
  const float* B_w    = (const float*)d_in[7];
  const float* B_b    = (const float*)d_in[8];
  const float* dl_w   = (const float*)d_in[9];
  const float* dl_b   = (const float*)d_in[10];
  const float* C_w    = (const float*)d_in[11];
  const float* C_b    = (const float*)d_in[12];
  const float* D_w    = (const float*)d_in[13];
  const float* D_b    = (const float*)d_in[14];
  const float* l2_w   = (const float*)d_in[15];
  const float* la_w   = (const float*)d_in[16];
  const float* ln_g   = (const float*)d_in[17];
  const float* ln_b   = (const float*)d_in[18];

  const int B = 2, L = 2048, V = 50257, d = 768, dh = 1536, nblk = 8;
  const int M = B * L;

  // ---- bf16 weight scratch inside d_out (head GEMM rewrites all of d_out) ----
  unsigned short* wscr = (unsigned short*)d_out;
  size_t wo = 0;
  auto walloc = [&](size_t n) { auto p = wscr + wo; wo += (n + 63) & ~(size_t)63; return p; };
  unsigned short* l1b = walloc((size_t)nblk * d  * d);
  unsigned short* awb = walloc((size_t)nblk * dh * d);
  unsigned short* bwb = walloc((size_t)nblk * dh * d);
  unsigned short* dwb = walloc((size_t)nblk * dh * d);
  unsigned short* cwb = walloc((size_t)nblk * d  * dh);
  unsigned short* Dwb = walloc((size_t)nblk * d  * d);
  unsigned short* l2b = walloc((size_t)nblk * d  * d);
  unsigned short* lab = walloc((size_t)nblk * d  * d);

  // ---- workspace: f32 temporaries + bf16 activations + bf16 embedding ----
  char* base = (char*)d_ws;
  size_t off = 0;
  auto fal = [&](size_t n) { float* p = (float*)(base + off); off += (n * 4 + 127) & ~(size_t)127; return p; };
  auto hal = [&](size_t n) { unsigned short* p = (unsigned short*)(base + off); off += (n * 2 + 127) & ~(size_t)127; return p; };
  float* h1f = fal((size_t)M * d);    // lin1 out (f32, conv input); reused as x_res
  float* dlt = fal((size_t)M * dh);
  float* atb = fal((size_t)M * dh);
  float* bxb = fal((size_t)M * dh);
  float* sf  = fal((size_t)M * d);    // s (f32); becomes residual x (f32) in place
  unsigned short* x_bf  = hal((size_t)M * d);
  unsigned short* h_bf  = hal((size_t)M * d);   // reused as layernorm out
  unsigned short* hs_bf = hal((size_t)M * dh);  // reused as t = s*x_res (M*d <= M*dh)
  unsigned short* s_bf  = hal((size_t)M * d);
  unsigned short* embbf = hal((size_t)V * d);
  unsigned short* t_bf  = hs_bf;
  unsigned short* ln_bf = h_bf;

  const int eb = 256;
  auto cvt = [&](const float* src, unsigned short* dst, size_t n) {
    size_t t = (n + 3) / 4;
    cvt_f32_bf16<<<(unsigned)((t + eb - 1) / eb), eb, 0, stream>>>(src, dst, n);
  };
  // one-time f32 -> bf16 of all GEMM operands
  cvt(lin1_w, l1b, (size_t)nblk * d  * d);
  cvt(A_w,    awb, (size_t)nblk * dh * d);
  cvt(B_w,    bwb, (size_t)nblk * dh * d);
  cvt(dl_w,   dwb, (size_t)nblk * dh * d);
  cvt(C_w,    cwb, (size_t)nblk * d  * dh);
  cvt(D_w,    Dwb, (size_t)nblk * d  * d);
  cvt(l2_w,   l2b, (size_t)nblk * d  * d);
  cvt(la_w,   lab, (size_t)nblk * d  * d);
  cvt(embedw, embbf, (size_t)V * d);

  const dim3 tb(GEMM_THREADS);
  auto gg = [&](int N_, int M_) { return dim3((unsigned)((N_ + 63) / 64), (unsigned)((M_ + 63) / 64)); };

  embed_gather_bf<<<(M * d + eb - 1) / eb, eb, 0, stream>>>(tokens, embbf, x_bf, M, d);

  for (int i = 0; i < nblk; ++i) {
    // h_pre = x @ lin1^T  (f32 for conv)
    gemm_bf16<0,0,false,false><<<gg(d, M), tb, 0, stream>>>(
        x_bf, l1b + (size_t)i * d * d, nullptr, nullptr, h1f, nullptr, M, d, d);
    // h = silu(conv(h_pre) + cb)  -> bf16
    conv_silu<<<(M * d + eb - 1) / eb, eb, 0, stream>>>(
        h1f, conv_w + (size_t)i * d * 3, conv_b + (size_t)i * d, h_bf, L, d, M * d);
    // delta = softplus(h@dw + db); A_t = h@aw + ab; B_x = h@bw + bb  (f32)
    gemm_bf16<0,2,true,false><<<gg(dh, M), tb, 0, stream>>>(
        h_bf, dwb + (size_t)i * dh * d, dl_b + (size_t)i * dh, nullptr, dlt, nullptr, M, dh, d);
    gemm_bf16<0,0,true,false><<<gg(dh, M), tb, 0, stream>>>(
        h_bf, awb + (size_t)i * dh * d, A_b + (size_t)i * dh, nullptr, atb, nullptr, M, dh, d);
    gemm_bf16<0,0,true,false><<<gg(dh, M), tb, 0, stream>>>(
        h_bf, bwb + (size_t)i * dh * d, B_b + (size_t)i * dh, nullptr, bxb, nullptr, M, dh, d);
    // sequential SSM scan -> bf16 hs
    ssm_scan<<<(B * dh + eb - 1) / eb, eb, 0, stream>>>(dlt, atb, bxb, hs_bf, B, L, dh);
    // s = hs@Cw + Cb  (f32), then s += h@Dw + Db  (dual f32 + bf16)
    gemm_bf16<0,0,true,false><<<gg(d, M), tb, 0, stream>>>(
        hs_bf, cwb + (size_t)i * d * dh, C_b + (size_t)i * d, nullptr, sf, nullptr, M, d, dh);
    gemm_bf16<2,0,true,true><<<gg(d, M), tb, 0, stream>>>(
        h_bf, Dwb + (size_t)i * d * d, D_b + (size_t)i * d, sf, sf, s_bf, M, d, d);
    // x_res = silu(s@l2)  (f32), t = s*x_res (bf16), x = s + t@last (f32 + bf16)
    gemm_bf16<0,1,false,false><<<gg(d, M), tb, 0, stream>>>(
        s_bf, l2b + (size_t)i * d * d, nullptr, nullptr, h1f, nullptr, M, d, d);
    mul_ew_bf<<<(M * d + eb - 1) / eb, eb, 0, stream>>>(sf, h1f, t_bf, M * d);
    gemm_bf16<2,0,false,true><<<gg(d, M), tb, 0, stream>>>(
        t_bf, lab + (size_t)i * d * d, nullptr, sf, sf, x_bf, M, d, d);
  }

  // LayerNorm (f32 residual in sf) -> bf16, then tied-embedding head -> d_out
  layernorm_bf<<<M, 256, 0, stream>>>(sf, ln_g, ln_b, ln_bf, d);
  gemm_bf16<0,0,false,false><<<gg(V, M), tb, 0, stream>>>(
      ln_bf, embbf, nullptr, nullptr, (float*)d_out, nullptr, M, V, d);
}
